// Seq2Seq_7842610283150
// MI455X (gfx1250) — compile-verified, matrix operand order checked by reading, not run on hardware
//
#include <hip/hip_runtime.h>
#include <math.h>

// ---------------- model dims ----------------
#define Hh     512
#define Ee     300
#define VOCAB  9488
#define Bb     32
#define Kk     2
#define Tt     64
#define BK     64          // Bb*Kk
#define G3H    1536        // 3*Hh
#define EP     320         // Ee padded to mult of 32 (16B aligned bf16 rows)
#define CXC    832         // (Hh+Ee)=812 padded
#define KT     128         // Kk*Tt
#define NEGV   -1e10f

// ---------------- WMMA types ----------------
typedef __attribute__((ext_vector_type(8)))  __bf16 v8bf;
typedef __attribute__((ext_vector_type(16))) __bf16 v16bf;
typedef __attribute__((ext_vector_type(8)))  float  v8f;

__device__ inline unsigned short bf16bits(float f) {
  unsigned u = __builtin_bit_cast(unsigned, f);
  u += 0x7fffu + ((u >> 16) & 1u);          // round-to-nearest-even
  return (unsigned short)(u >> 16);
}

__device__ inline v16bf load_frag16(const unsigned short* p) {
  // 16B-aligned: two b128 loads -> 16 bf16 (K sub-block lo 0..7, hi 16..23 pattern)
  v8bf lo = __builtin_bit_cast(v8bf, *reinterpret_cast<const uint4*>(p));
  v8bf hi = __builtin_bit_cast(v8bf, *reinterpret_cast<const uint4*>(p + 16));
  return __builtin_shufflevector(lo, hi, 0,1,2,3,4,5,6,7,8,9,10,11,12,13,14,15);
}

// ---------------- bf16 NT GEMM: C[m,n] = sum_k A[m,k]*W[n,k] (+bias[n]) ----------------
// One wave32 computes a 32x32 C patch (2x2 tiles of 16x16) via
// v_wmma_f32_16x16x32_bf16: 4 WMMAs per 4 b128-pair loads -> 2x arithmetic
// intensity vs 1 tile/wave. M must be a multiple of 32 (true for 4096/64/32);
// odd N tile counts handled by clamping the 2nd N tile and guarding its store.
__global__ __launch_bounds__(256)
void wmma_gemm_nt(const unsigned short* __restrict__ A, int lda,
                  const unsigned short* __restrict__ W, int ldb,
                  const float* __restrict__ bias,
                  float* __restrict__ C, int ldc,
                  int M, int N, int K, int relu)
{
  int wave = (blockIdx.x * blockDim.x + threadIdx.x) >> 5;
  int ntn = N >> 4;              // 16-wide N tiles
  int ng  = (ntn + 1) >> 1;      // groups of 2 N tiles
  int mg  = M >> 5;              // groups of 2 M tiles (M % 32 == 0)
  if (wave >= mg * ng) return;
  int mt  = (wave / ng) * 2;
  int nt0 = (wave % ng) * 2;
  int nt1 = nt0 + 1;
  int nt1c = (nt1 < ntn) ? nt1 : (ntn - 1);   // clamped (duplicate compute, store guarded)
  int lane = threadIdx.x & 31;
  int half = lane >> 4;          // K sub-block select
  int r    = lane & 15;          // row within tile
  const unsigned short* arow0 = A + (size_t)(mt * 16 + r) * lda + half * 8;
  const unsigned short* arow1 = arow0 + (size_t)16 * lda;
  const unsigned short* brow0 = W + (size_t)(nt0 * 16 + r) * ldb + half * 8;
  const unsigned short* brow1 = W + (size_t)(nt1c * 16 + r) * ldb + half * 8;
  v8f acc00 = {}, acc01 = {}, acc10 = {}, acc11 = {};
  for (int k0 = 0; k0 < K; k0 += 32) {
    v16bf a0 = load_frag16(arow0 + k0);
    v16bf a1 = load_frag16(arow1 + k0);
    v16bf b0 = load_frag16(brow0 + k0);
    v16bf b1 = load_frag16(brow1 + k0);
    if (k0 + 32 < K) {
      __builtin_prefetch(arow0 + k0 + 32, 0, 3);
      __builtin_prefetch(arow1 + k0 + 32, 0, 3);
      __builtin_prefetch(brow0 + k0 + 32, 0, 3);
      __builtin_prefetch(brow1 + k0 + 32, 0, 3);
    }
    acc00 = __builtin_amdgcn_wmma_f32_16x16x32_bf16(false, a0, false, b0, (short)0, acc00, false, false);
    acc01 = __builtin_amdgcn_wmma_f32_16x16x32_bf16(false, a0, false, b1, (short)0, acc01, false, false);
    acc10 = __builtin_amdgcn_wmma_f32_16x16x32_bf16(false, a1, false, b0, (short)0, acc10, false, false);
    acc11 = __builtin_amdgcn_wmma_f32_16x16x32_bf16(false, a1, false, b1, (short)0, acc11, false, false);
  }
  int n0 = nt0 * 16 + r;
  int n1 = nt1c * 16 + r;
  float bv0 = bias ? bias[n0] : 0.0f;
  float bv1 = bias ? bias[n1] : 0.0f;
  int store1 = (nt1 < ntn);
#pragma unroll
  for (int v = 0; v < 8; ++v) {
    int m0 = mt * 16 + v + half * 8;
    int m1 = m0 + 16;
    float v00 = acc00[v] + bv0, v01 = acc01[v] + bv1;
    float v10 = acc10[v] + bv0, v11 = acc11[v] + bv1;
    if (relu) {
      v00 = fmaxf(v00, 0.0f); v01 = fmaxf(v01, 0.0f);
      v10 = fmaxf(v10, 0.0f); v11 = fmaxf(v11, 0.0f);
    }
    C[(size_t)m0 * ldc + n0] = v00;
    C[(size_t)m1 * ldc + n0] = v10;
    if (store1) {
      C[(size_t)m0 * ldc + n1] = v01;
      C[(size_t)m1 * ldc + n1] = v11;
    }
  }
}

// ---------------- float -> bf16 with row/col zero-padding ----------------
__global__ void f2bf_pad(const float* __restrict__ src, int srcld, int src_rows, int cols,
                         unsigned short* __restrict__ dst, int dstld, int dst_rows)
{
  size_t i = (size_t)blockIdx.x * blockDim.x + threadIdx.x;
  size_t tot = (size_t)dst_rows * dstld;
  if (i >= tot) return;
  int c = (int)(i % dstld);
  int r = (int)(i / dstld);
  float v = (r < src_rows && c < cols) ? src[(size_t)r * srcld + c] : 0.0f;
  dst[i] = bf16bits(v);
}

__global__ void pad_bias_ep(const float* __restrict__ b, float* __restrict__ dst) {
  int i = blockIdx.x * blockDim.x + threadIdx.x;
  if (i < EP) dst[i] = (i < Ee) ? b[i] : 0.0f;
}

// ---------------- embedding gather -> xs_bf (T*BK, EP) ----------------
__global__ void gather_src_emb(const int* __restrict__ src, const float* __restrict__ emb,
                               unsigned short* __restrict__ xs_bf)
{
  size_t i = (size_t)blockIdx.x * blockDim.x + threadIdx.x;
  if (i >= (size_t)Tt * BK * EP) return;
  int c = (int)(i % EP);
  size_t row = i / EP;
  int bk = (int)(row % BK);
  int t  = (int)(row / BK);
  int tok = src[bk * Tt + t];
  float v = (c < Ee) ? emb[(size_t)tok * Ee + c] : 0.0f;
  xs_bf[i] = bf16bits(v);
}

__global__ void src_len_k(const int* __restrict__ src, int* __restrict__ slen) {
  int bk = threadIdx.x;
  if (bk >= BK) return;
  int last = 0;
  for (int t = 0; t < Tt; ++t) if (src[bk * Tt + t] != 0) last = t;
  slen[bk] = last;
}

__global__ void init_h_k(const float* __restrict__ h0v, float* __restrict__ h,
                         unsigned short* __restrict__ h_bf) {
  int i = blockIdx.x * blockDim.x + threadIdx.x;
  if (i >= BK * Hh) return;
  float v = h0v[i & (Hh - 1)];
  h[i] = v; h_bf[i] = bf16bits(v);
}

// ---------------- fused LayerNorm-GRU step ----------------
// gi,gh: (rows,1536) pre-activations (bias already added by GEMM epilogue)
__global__ __launch_bounds__(256)
void ln_gru_step(const float* __restrict__ gi, const float* __restrict__ gh,
                 const float* __restrict__ g_i, const float* __restrict__ b_i,
                 const float* __restrict__ g_h, const float* __restrict__ b_h,
                 float* __restrict__ h, unsigned short* __restrict__ h_bf,
                 float* __restrict__ out_extra, long long out_rstride)
{
  int row = blockIdx.x, tid = threadIdx.x;
  const float* gir = gi + (size_t)row * G3H;
  const float* ghr = gh + (size_t)row * G3H;
  __shared__ float s[4][256];
  float a0 = 0, a1 = 0, a2 = 0, a3 = 0;
  for (int j = tid; j < G3H; j += 256) {
    float x = gir[j], y = ghr[j];
    a0 += x; a1 += x * x; a2 += y; a3 += y * y;
  }
  s[0][tid] = a0; s[1][tid] = a1; s[2][tid] = a2; s[3][tid] = a3;
  __syncthreads();
  for (int o = 128; o; o >>= 1) {
    if (tid < o) {
      s[0][tid] += s[0][tid + o]; s[1][tid] += s[1][tid + o];
      s[2][tid] += s[2][tid + o]; s[3][tid] += s[3][tid + o];
    }
    __syncthreads();
  }
  float mi = s[0][0] * (1.0f / G3H);
  float vi = s[1][0] * (1.0f / G3H) - mi * mi;
  float mh = s[2][0] * (1.0f / G3H);
  float vh = s[3][0] * (1.0f / G3H) - mh * mh;
  float ri = rsqrtf(vi + 1e-5f), rh = rsqrtf(vh + 1e-5f);
  for (int j = tid; j < Hh; j += 256) {
    float ir  = (gir[j]           - mi) * ri * g_i[j]           + b_i[j];
    float iz  = (gir[j + Hh]      - mi) * ri * g_i[j + Hh]      + b_i[j + Hh];
    float inn = (gir[j + 2 * Hh]  - mi) * ri * g_i[j + 2 * Hh]  + b_i[j + 2 * Hh];
    float hr  = (ghr[j]           - mh) * rh * g_h[j]           + b_h[j];
    float hz  = (ghr[j + Hh]      - mh) * rh * g_h[j + Hh]      + b_h[j + Hh];
    float hn  = (ghr[j + 2 * Hh]  - mh) * rh * g_h[j + 2 * Hh]  + b_h[j + 2 * Hh];
    float rg = 1.0f / (1.0f + __expf(-(ir + hr)));
    float zg = 1.0f / (1.0f + __expf(-(iz + hz)));
    float ng = tanhf(inn + rg * hn);
    size_t hi = (size_t)row * Hh + j;
    float hnew = (1.0f - zg) * ng + zg * h[hi];
    h[hi] = hnew;
    if (h_bf) h_bf[hi] = bf16bits(hnew);
    if (out_extra) out_extra[(size_t)row * out_rstride + j] = hnew;
  }
}

// ---------------- encoder combine ----------------
__global__ void enc_combine(const float* __restrict__ o0, const float* __restrict__ o1,
                            float* __restrict__ enc, unsigned short* __restrict__ enc_bf)
{
  size_t i = (size_t)blockIdx.x * blockDim.x + threadIdx.x;
  if (i >= (size_t)BK * Tt * Hh) return;
  int hh = (int)(i % Hh);
  size_t rem = i / Hh;
  int t = (int)(rem % Tt);
  int bk = (int)(rem / Tt);
  size_t o = ((size_t)(t * BK + bk) * 2) * Hh + hh;  // (t,bk,dir,h)
  float v = 0.25f * (o0[o] + o0[o + Hh] + o1[o] + o1[o + Hh]);
  enc[i] = v; enc_bf[i] = bf16bits(v);
}

__global__ void enc_hidden_k(const float* __restrict__ o0, const float* __restrict__ o1,
                             const int* __restrict__ slen, float* __restrict__ eh)
{
  int i = blockIdx.x * blockDim.x + threadIdx.x;
  if (i >= BK * Hh) return;
  int hh = i & (Hh - 1), bk = i >> 9;
  int sl = slen[bk];
  size_t ob = ((size_t)(0 * BK + bk) * 2 + 1) * Hh + hh;   // t=0, backward
  size_t of = ((size_t)(sl * BK + bk) * 2 + 0) * Hh + hh;  // t=src_len, forward
  eh[i] = 0.25f * (o0[ob] + o1[ob] + o0[of] + o1[of]);
}

__global__ void hidden0_k(const float* __restrict__ eh, float* __restrict__ hid,
                          unsigned short* __restrict__ hid_bf)
{
  int i = blockIdx.x * blockDim.x + threadIdx.x;
  if (i >= Bb * Hh) return;
  int hh = i & (Hh - 1), b = i >> 9;
  float v = 0.5f * (eh[(size_t)(2 * b) * Hh + hh] + eh[(size_t)(2 * b + 1) * Hh + hh]);
  hid[i] = v; hid_bf[i] = bf16bits(v);
}

__global__ void init_tok_k(int* tok) { if (threadIdx.x < Bb) tok[threadIdx.x] = 0; }

// ---------------- decoder: attention scores ----------------
__global__ __launch_bounds__(128)
void att_scores(const float* __restrict__ feat, const float* __restrict__ keys,
                const float* __restrict__ vvec, const int* __restrict__ src,
                float* __restrict__ esc)
{
  int b = blockIdx.x >> 7, n = blockIdx.x & 127, tid = threadIdx.x;
  const float* fb = feat + (size_t)b * Hh;
  const float* kb = keys + (size_t)(b * KT + n) * Hh;
  float acc = 0;
  for (int j = tid; j < Hh; j += 128) acc += tanhf(fb[j] + kb[j]) * vvec[j];
  __shared__ float sh[128];
  sh[tid] = acc; __syncthreads();
  for (int o = 64; o; o >>= 1) { if (tid < o) sh[tid] += sh[tid + o]; __syncthreads(); }
  if (tid == 0) esc[b * KT + n] = (src[b * KT + n] == 0) ? NEGV : sh[0];
}

__global__ __launch_bounds__(128)
void softmax128(const float* __restrict__ e, float* __restrict__ a)
{
  int b = blockIdx.x, n = threadIdx.x;
  __shared__ float sh[128];
  float x = e[b * KT + n];
  sh[n] = x; __syncthreads();
  for (int o = 64; o; o >>= 1) { if (n < o) sh[n] = fmaxf(sh[n], sh[n + o]); __syncthreads(); }
  float mx = sh[0]; __syncthreads();
  float ex = __expf(x - mx);
  sh[n] = ex; __syncthreads();
  for (int o = 64; o; o >>= 1) { if (n < o) sh[n] += sh[n + o]; __syncthreads(); }
  a[b * KT + n] = ex / sh[0];
}

__global__ __launch_bounds__(256)
void ctx_k(const float* __restrict__ a, const float* __restrict__ enc,
           float* __restrict__ ctx, unsigned short* __restrict__ cat_xc)
{
  int b = blockIdx.x, tid = threadIdx.x;
  for (int hh = tid; hh < Hh; hh += 256) {
    float acc = 0;
    for (int n = 0; n < KT; ++n)
      acc += a[b * KT + n] * enc[(size_t)(b * KT + n) * Hh + hh];
    ctx[(size_t)b * Hh + hh] = acc;
    cat_xc[(size_t)b * CXC + Ee + hh] = bf16bits(acc);   // cols 300..811
  }
}

__global__ void fill_ein(const float* __restrict__ emb, const int* __restrict__ tok,
                         unsigned short* __restrict__ cat_xc)
{
  int b = blockIdx.x, tid = threadIdx.x;
  int tk = tok[b];
  for (int c = tid; c < Ee; c += 256)
    cat_xc[(size_t)b * CXC + c] = bf16bits(emb[(size_t)tk * Ee + c]);
  for (int c = (Hh + Ee) + tid; c < CXC; c += 256)
    cat_xc[(size_t)b * CXC + c] = 0;
}

__global__ __launch_bounds__(256)
void pg_k(const float* __restrict__ hid, const float* __restrict__ xv,
          const float* __restrict__ Wpg, const float* __restrict__ bpg,
          float* __restrict__ pg)
{
  int b = blockIdx.x, tid = threadIdx.x;
  float acc = 0;
  for (int j = tid; j < Hh; j += 256) acc += hid[(size_t)b * Hh + j] * Wpg[j];
  for (int j = tid; j < Ee; j += 256) acc += xv[(size_t)b * EP + j] * Wpg[Hh + j];
  __shared__ float sh[256];
  sh[tid] = acc; __syncthreads();
  for (int o = 128; o; o >>= 1) { if (tid < o) sh[tid] += sh[tid + o]; __syncthreads(); }
  if (tid == 0) pg[b] = 1.0f / (1.0f + __expf(-(sh[0] + bpg[0])));
}

__global__ void cat_ho_k(const float* __restrict__ h2, const float* __restrict__ ctx,
                         unsigned short* __restrict__ cat_ho)
{
  int b = blockIdx.x, tid = threadIdx.x;
  for (int j = tid; j < Hh; j += 256) {
    cat_ho[(size_t)b * 1024 + j]      = bf16bits(h2[(size_t)b * Hh + j]);
    cat_ho[(size_t)b * 1024 + Hh + j] = bf16bits(ctx[(size_t)b * Hh + j]);
  }
}

__global__ __launch_bounds__(256)
void softmax_v(float* __restrict__ out, const float* __restrict__ pg)
{
  int b = blockIdx.x, tid = threadIdx.x;
  float* row = out + (size_t)b * VOCAB;
  __shared__ float sh[256];
  float mx = -INFINITY;
  for (int v = tid; v < VOCAB; v += 256) mx = fmaxf(mx, row[v]);
  sh[tid] = mx; __syncthreads();
  for (int o = 128; o; o >>= 1) { if (tid < o) sh[tid] = fmaxf(sh[tid], sh[tid + o]); __syncthreads(); }
  mx = sh[0]; __syncthreads();
  float s = 0;
  for (int v = tid; v < VOCAB; v += 256) s += __expf(row[v] - mx);
  sh[tid] = s; __syncthreads();
  for (int o = 128; o; o >>= 1) { if (tid < o) sh[tid] += sh[tid + o]; __syncthreads(); }
  float inv = pg[b] / sh[0];
  for (int v = tid; v < VOCAB; v += 256) row[v] = __expf(row[v] - mx) * inv;
}

__global__ __launch_bounds__(128)
void scatter_copy(float* __restrict__ out, const int* __restrict__ src,
                  const float* __restrict__ a, const float* __restrict__ pg)
{
  int b = blockIdx.x, n = threadIdx.x;
  float w = (1.0f - pg[b]) * a[b * KT + n];
  atomicAdd(&out[(size_t)b * VOCAB + src[b * KT + n]], w);
}

__global__ __launch_bounds__(256)
void argmax_log(float* __restrict__ out, int* __restrict__ tok)
{
  int b = blockIdx.x, tid = threadIdx.x;
  float* row = out + (size_t)b * VOCAB;
  __shared__ float sv[256]; __shared__ int si[256];
  float best = -INFINITY; int bi = 0;
  for (int v = tid; v < VOCAB; v += 256) {
    float x = row[v];
    if (x > best) { best = x; bi = v; }
  }
  sv[tid] = best; si[tid] = bi; __syncthreads();
  for (int o = 128; o; o >>= 1) {
    if (tid < o) {
      if (sv[tid + o] > sv[tid] || (sv[tid + o] == sv[tid] && si[tid + o] < si[tid])) {
        sv[tid] = sv[tid + o]; si[tid] = si[tid + o];
      }
    }
    __syncthreads();
  }
  if (tid == 0) tok[b] = si[0];
  __syncthreads();
  for (int v = tid; v < VOCAB; v += 256) row[v] = logf(row[v]);
}

// ======================= host orchestration =======================
struct CellP { const float *Whh, *Wih, *bh, *bhh, *bi, *bih, *gh, *gi; };

static void gemm(const unsigned short* A, int lda, const unsigned short* W, int ldb,
                 const float* bias, float* C, int ldc, int M, int N, int K,
                 int relu, hipStream_t s)
{
  int ntn = N >> 4;
  int groups = (M >> 5) * ((ntn + 1) >> 1);    // one wave per 32x32 patch
  int blocks = (groups + 7) >> 3;
  wmma_gemm_nt<<<blocks, 256, 0, s>>>(A, lda, W, ldb, bias, C, ldc, M, N, K, relu);
}

static void conv(const float* src, int srcld, int srows, int cols,
                 unsigned short* dst, int dstld, int drows, hipStream_t s)
{
  size_t tot = (size_t)drows * dstld;
  f2bf_pad<<<(unsigned)((tot + 255) / 256), 256, 0, s>>>(src, srcld, srows, cols, dst, dstld, drows);
}

extern "C" void kernel_launch(void* const* d_in, const int* in_sizes, int n_in,
                              void* d_out, int out_size, void* d_ws, size_t ws_size,
                              hipStream_t stream)
{
  (void)in_sizes; (void)n_in; (void)out_size; (void)ws_size;
  auto F = [&](int i) { return (const float*)d_in[i]; };
  // JAX pytree (sorted dict keys) flattening of setup_inputs():
  // 0:src 1:tgt_idx 2:Wh 3:attn_W 4:attn_b 5:Wout 6:Wpg 7:Wxc 8:bout 9:bpg 10:bxc
  // 11-18: dec.cell{Whh,Wih,bh,bhh,bi,bih,gh,gi} 19:emb 20:enc.h0
  // 21-28:l0b 29-36:l0f 37-44:l1b 45-52:l1f (same cell order) 53:v
  const int* src = (const int*)d_in[0];
  const float *Wh = F(2), *attnW = F(3), *attn_b = F(4);
  const float *Wout = F(5), *Wpg = F(6), *Wxc = F(7), *bout = F(8), *bpg = F(9), *bxc = F(10);
  CellP dc  = {F(11), F(12), F(13), F(14), F(15), F(16), F(17), F(18)};
  const float* emb = F(19);
  const float* h0  = F(20);
  CellP l0b = {F(21), F(22), F(23), F(24), F(25), F(26), F(27), F(28)};
  CellP l0f = {F(29), F(30), F(31), F(32), F(33), F(34), F(35), F(36)};
  CellP l1b = {F(37), F(38), F(39), F(40), F(41), F(42), F(43), F(44)};
  CellP l1f = {F(45), F(46), F(47), F(48), F(49), F(50), F(51), F(52)};
  const float* vvec = F(53);

  // ---- workspace carve (256B aligned) ----
  char* wp = (char*)d_ws;
  auto alloc = [&](size_t bytes) { void* r = wp; wp += (bytes + 255) & ~(size_t)255; return r; };
  auto aU = [&](size_t n) { return (unsigned short*)alloc(n * 2); };
  auto aF = [&](size_t n) { return (float*)alloc(n * 4); };
  auto aI = [&](size_t n) { return (int*)alloc(n * 4); };

  unsigned short* Wih0f_bf = aU((size_t)G3H * EP);
  unsigned short* Wih0b_bf = aU((size_t)G3H * EP);
  unsigned short* Wih1f_bf = aU((size_t)G3H * 1024);
  unsigned short* Wih1b_bf = aU((size_t)G3H * 1024);
  unsigned short* Whh0f_bf = aU((size_t)G3H * Hh);
  unsigned short* Whh0b_bf = aU((size_t)G3H * Hh);
  unsigned short* Whh1f_bf = aU((size_t)G3H * Hh);
  unsigned short* Whh1b_bf = aU((size_t)G3H * Hh);
  unsigned short* Whhd_bf  = aU((size_t)G3H * Hh);
  unsigned short* Wihd_bf  = aU((size_t)G3H * EP);
  unsigned short* Wh_bf    = aU((size_t)Hh * Hh);
  unsigned short* attnW_bf = aU((size_t)Hh * Hh);
  unsigned short* Wxc_bf   = aU((size_t)EP * CXC);
  unsigned short* Wout_bf  = aU((size_t)VOCAB * 1024);
  unsigned short* xs_bf    = aU((size_t)Tt * BK * EP);
  unsigned short* inp1_bf  = aU((size_t)Tt * BK * 1024);
  unsigned short* h_bf     = aU((size_t)BK * Hh);
  unsigned short* enc_bf   = aU((size_t)BK * Tt * Hh);
  unsigned short* hid_bf   = aU((size_t)Bb * Hh);
  unsigned short* catxc_bf = aU((size_t)Bb * CXC);
  unsigned short* x_bf     = aU((size_t)Bb * EP);
  unsigned short* catho_bf = aU((size_t)Bb * 1024);

  float* gi_all   = aF((size_t)Tt * BK * G3H);
  float* gh_buf   = aF((size_t)BK * G3H);
  float* h_f      = aF((size_t)BK * Hh);
  float* out0     = aF((size_t)Tt * BK * 2 * Hh);
  float* out1     = aF((size_t)Tt * BK * 2 * Hh);
  float* enc      = aF((size_t)BK * Tt * Hh);
  float* keys     = aF((size_t)BK * Tt * Hh);
  float* enc_hid  = aF((size_t)BK * Hh);
  float* hidden   = aF((size_t)Bb * Hh);
  float* feat     = aF((size_t)Bb * Hh);
  float* escore   = aF((size_t)Bb * KT);
  float* attw     = aF((size_t)Bb * KT);
  float* ctx      = aF((size_t)Bb * Hh);
  float* xvec     = aF((size_t)Bb * EP);
  float* pgv      = aF(Bb);
  float* gi_dec   = aF((size_t)Bb * G3H);
  float* gh_dec   = aF((size_t)Bb * G3H);
  float* bxc_pad  = aF(EP);
  int*   slen     = aI(BK);
  int*   tok      = aI(Bb);

  // ---- weight conversions to bf16 (padded) ----
  conv(l0f.Wih, Ee, G3H, Ee, Wih0f_bf, EP, G3H, stream);
  conv(l0b.Wih, Ee, G3H, Ee, Wih0b_bf, EP, G3H, stream);
  conv(l1f.Wih, 1024, G3H, 1024, Wih1f_bf, 1024, G3H, stream);
  conv(l1b.Wih, 1024, G3H, 1024, Wih1b_bf, 1024, G3H, stream);
  conv(l0f.Whh, Hh, G3H, Hh, Whh0f_bf, Hh, G3H, stream);
  conv(l0b.Whh, Hh, G3H, Hh, Whh0b_bf, Hh, G3H, stream);
  conv(l1f.Whh, Hh, G3H, Hh, Whh1f_bf, Hh, G3H, stream);
  conv(l1b.Whh, Hh, G3H, Hh, Whh1b_bf, Hh, G3H, stream);
  conv(dc.Whh,  Hh, G3H, Hh, Whhd_bf,  Hh, G3H, stream);
  conv(dc.Wih,  Ee, G3H, Ee, Wihd_bf,  EP, G3H, stream);
  conv(Wh,    Hh, Hh, Hh, Wh_bf,    Hh, Hh, stream);
  conv(attnW, Hh, Hh, Hh, attnW_bf, Hh, Hh, stream);
  conv(Wxc, Hh + Ee, Ee, Hh + Ee, Wxc_bf, CXC, EP, stream);   // pad rows 300->320, cols 812->832
  conv(Wout, 1024, VOCAB, 1024, Wout_bf, 1024, VOCAB, stream);
  pad_bias_ep<<<2, 256, 0, stream>>>(bxc, bxc_pad);

  // ---- encoder inputs ----
  gather_src_emb<<<(unsigned)(((size_t)Tt * BK * EP + 255) / 256), 256, 0, stream>>>(src, emb, xs_bf);
  src_len_k<<<1, 64, 0, stream>>>(src, slen);

  // ---- 2-layer bidirectional LayerNorm-GRU encoder ----
  const CellP cells[4] = {l0f, l0b, l1f, l1b};
  const unsigned short* cWih[4] = {Wih0f_bf, Wih0b_bf, Wih1f_bf, Wih1b_bf};
  const unsigned short* cWhh[4] = {Whh0f_bf, Whh0b_bf, Whh1f_bf, Whh1b_bf};
  for (int c = 0; c < 4; ++c) {
    int l = c >> 1, d = c & 1;
    if (c == 2)   // layer-1 input = out0 reshaped (T*BK, 1024) -> bf16
      conv(out0, 1024, Tt * BK, 1024, inp1_bf, 1024, Tt * BK, stream);
    const unsigned short* X = (l == 0) ? xs_bf : inp1_bf;
    int ldx = (l == 0) ? EP : 1024;
    // gi for all timesteps: big batched GEMM
    gemm(X, ldx, cWih[c], ldx, cells[c].bih, gi_all, G3H, Tt * BK, G3H, ldx, 0, stream);
    init_h_k<<<(BK * Hh + 255) / 256, 256, 0, stream>>>(h0 + (size_t)(l * 2 + d) * Hh, h_f, h_bf);
    float* outL = (l == 0) ? out0 : out1;
    for (int s = 0; s < Tt; ++s) {
      int t = (d == 0) ? s : (Tt - 1 - s);
      gemm(h_bf, Hh, cWhh[c], Hh, cells[c].bhh, gh_buf, G3H, BK, G3H, Hh, 0, stream);
      ln_gru_step<<<BK, 256, 0, stream>>>(gi_all + (size_t)t * BK * G3H, gh_buf,
                                          cells[c].gi, cells[c].bi, cells[c].gh, cells[c].bh,
                                          h_f, h_bf,
                                          outL + ((size_t)t * BK * 2 + d) * Hh, (long long)(2 * Hh));
    }
  }

  // ---- encoder post-processing ----
  enc_combine<<<(unsigned)(((size_t)BK * Tt * Hh + 255) / 256), 256, 0, stream>>>(out0, out1, enc, enc_bf);
  enc_hidden_k<<<(BK * Hh + 255) / 256, 256, 0, stream>>>(out0, out1, slen, enc_hid);
  hidden0_k<<<(Bb * Hh + 255) / 256, 256, 0, stream>>>(enc_hid, hidden, hid_bf);
  gemm(enc_bf, Hh, Wh_bf, Hh, nullptr, keys, Hh, Tt * BK, Hh, Hh, 0, stream);
  init_tok_k<<<1, 32, 0, stream>>>(tok);

  // ---- decoder: 64 steps ----
  for (int t = 0; t < Tt; ++t) {
    float* outrow = (float*)d_out + (size_t)t * Bb * VOCAB;
    gemm(hid_bf, Hh, attnW_bf, Hh, attn_b, feat, Hh, Bb, Hh, Hh, 0, stream);
    att_scores<<<Bb * KT, 128, 0, stream>>>(feat, keys, vvec, src, escore);
    softmax128<<<Bb, 128, 0, stream>>>(escore, attw);
    ctx_k<<<Bb, 256, 0, stream>>>(attw, enc, ctx, catxc_bf);
    fill_ein<<<Bb, 256, 0, stream>>>(emb, tok, catxc_bf);
    gemm(catxc_bf, CXC, Wxc_bf, CXC, bxc_pad, xvec, EP, Bb, EP, CXC, 1, stream);
    conv(xvec, EP, Bb, EP, x_bf, EP, Bb, stream);
    pg_k<<<Bb, 256, 0, stream>>>(hidden, xvec, Wpg, bpg, pgv);
    gemm(x_bf, EP, Wihd_bf, EP, dc.bih, gi_dec, G3H, Bb, G3H, EP, 0, stream);
    gemm(hid_bf, Hh, Whhd_bf, Hh, dc.bhh, gh_dec, G3H, Bb, G3H, Hh, 0, stream);
    ln_gru_step<<<Bb, 256, 0, stream>>>(gi_dec, gh_dec, dc.gi, dc.bi, dc.gh, dc.bh,
                                        hidden, hid_bf, nullptr, 0);
    cat_ho_k<<<Bb, 256, 0, stream>>>(hidden, ctx, catho_bf);
    gemm(catho_bf, 1024, Wout_bf, 1024, bout, outrow, VOCAB, Bb, VOCAB, 1024, 0, stream);
    softmax_v<<<Bb, 256, 0, stream>>>(outrow, pgv);
    scatter_copy<<<Bb, 128, 0, stream>>>(outrow, src, attw, pgv);
    argmax_log<<<Bb, 256, 0, stream>>>(outrow, tok);
  }
}
